// MultiHeadLatentAttention_71416716198097
// MI455X (gfx1250) — matrix operand order repaired and checked by reading, not compiled, not run on hardware
//
#include <hip/hip_runtime.h>

// ---------------- types ----------------
typedef __bf16 bf16_t;
typedef __attribute__((ext_vector_type(16))) __bf16 v16bf;
typedef __attribute__((ext_vector_type(8)))  __bf16 v8bf;
typedef __attribute__((ext_vector_type(8)))  float  v8f;
typedef int vec4i __attribute__((vector_size(16)));

static constexpr int Bb = 2, Tt = 2048, Dd = 2048;
static constexpr int Hh = 16, HD = 128, Rr = 64;
static constexpr int CKV = 512, CQ = 1536;
static constexpr int QKD = HD + Rr; // 192

// ---------------- async global->LDS copy (CDNA5 ASYNCcnt path) ----------------
#if __has_builtin(__builtin_amdgcn_global_load_async_to_lds_b128) && \
    __has_builtin(__builtin_amdgcn_s_wait_asynccnt)
#define HAVE_ASYNC_LDS 1
#else
#define HAVE_ASYNC_LDS 0
#endif

__device__ __forceinline__ void async_copy16(const bf16_t* __restrict__ g,
                                             bf16_t* __restrict__ l) {
#if HAVE_ASYNC_LDS
  __builtin_amdgcn_global_load_async_to_lds_b128(
      (__attribute__((address_space(1))) vec4i*)(g),
      (__attribute__((address_space(3))) vec4i*)(l), 0, 0);
#else
  *reinterpret_cast<v8bf*>(l) = *reinterpret_cast<const v8bf*>(g);
#endif
}

__device__ __forceinline__ void async_wait() {
#if HAVE_ASYNC_LDS
  __builtin_amdgcn_s_wait_asynccnt(0);
#endif
}

__device__ __forceinline__ v8f v8f_zero() {
  v8f z;
#pragma unroll
  for (int i = 0; i < 8; ++i) z[i] = 0.0f;
  return z;
}

__device__ __forceinline__ v8f wmma_bf16(v16bf a, v16bf b, v8f c) {
  return __builtin_amdgcn_wmma_f32_16x16x32_bf16(false, a, false, b, (short)0, c,
                                                 false, false);
}

// ---------------- elementwise kernels ----------------
__global__ void cvt_f32_bf16(const float* __restrict__ s, bf16_t* __restrict__ d, int n) {
  int i = blockIdx.x * blockDim.x + threadIdx.x;
  if (i < n) d[i] = (bf16_t)s[i];
}

// in-place rope on bf16 tensor shaped (B*T, Hn, 2*half); pos is (B*T, half) f32
__global__ void rope_bf16(bf16_t* __restrict__ x, const float* __restrict__ pos,
                          int Hn, int half, int n) {
  int idx = blockIdx.x * blockDim.x + threadIdx.x;
  if (idx >= n) return;
  int j   = idx % half;
  int h   = (idx / half) % Hn;
  int bt  = idx / (half * Hn);
  float p = pos[bt * half + j];
  float c = cosf(p), s = sinf(p);
  int base = (bt * Hn + h) * (2 * half);
  float x1 = (float)x[base + j];
  float x2 = (float)x[base + half + j];
  x[base + j]        = (bf16_t)(x1 * c - x2 * s);
  x[base + half + j] = (bf16_t)(x1 * s + x2 * c);
}

// q_full(B*T, H, 192) <- concat(q_content(B*T, H*128), q_rope(B*T, H*64))
__global__ void pack_q_kernel(const bf16_t* __restrict__ qc, const bf16_t* __restrict__ qr,
                              bf16_t* __restrict__ out, int n) {
  int idx = blockIdx.x * blockDim.x + threadIdx.x;
  if (idx >= n) return;
  int e = idx % QKD;
  int h = (idx / QKD) % Hh;
  int bt = idx / (QKD * Hh);
  out[idx] = (e < HD) ? qc[bt * Dd + h * HD + e]
                      : qr[(bt * Hh + h) * Rr + (e - HD)];
}

// k_full(B*T, H, 192) <- concat(full_k(B*T, H*128), broadcast k_rope(B*T, 64))
__global__ void pack_k_kernel(const bf16_t* __restrict__ fk, const bf16_t* __restrict__ kr,
                              bf16_t* __restrict__ out, int n) {
  int idx = blockIdx.x * blockDim.x + threadIdx.x;
  if (idx >= n) return;
  int e = idx % QKD;
  int h = (idx / QKD) % Hh;
  int bt = idx / (QKD * Hh);
  out[idx] = (e < HD) ? fk[bt * Dd + h * HD + e]
                      : kr[bt * Rr + (e - HD)];
}

// ---------------- WMMA GEMM: C[M,N] = A[M,K] @ W[N,K]^T ----------------
// A, W bf16 row-major. Block = 256 threads = 8 waves. Tile M=128, N=64, K-step 32.
template <bool BF16OUT>
__global__ __launch_bounds__(256, 1)
void gemm_wmma(const bf16_t* __restrict__ A, int lda,
               const bf16_t* __restrict__ W, int ldw,
               void* __restrict__ C, int ldc, int K) {
  __shared__ bf16_t aS[128 * 40]; // 32 K-cols, padded stride 40

  const int tid  = threadIdx.x;
  const int w    = tid >> 5;
  const int lane = tid & 31;
  const int rl   = lane & 15;
  const int hi   = lane >> 4;
  const int h8   = hi * 8;

  const int n0  = blockIdx.x * 64;
  const int m0  = blockIdx.y * 128;
  const int m0w = m0 + w * 16;

  v8f acc[4];
#pragma unroll
  for (int t = 0; t < 4; ++t) acc[t] = v8f_zero();

  const int ldrow = tid >> 1;          // 0..127
  const int ldhalf = (tid & 1) * 16;   // 0 or 16

  for (int k0 = 0; k0 < K; k0 += 32) {
    // cooperative A tile load: 128 x 32 bf16 via async global->LDS
    {
      const bf16_t* src = A + (size_t)(m0 + ldrow) * lda + k0 + ldhalf;
      bf16_t* dst = &aS[ldrow * 40 + ldhalf];
      async_copy16(src, dst);
      async_copy16(src + 8, dst + 8);
      if (k0 + 32 < K) __builtin_prefetch(src + 32, 0, 0);
    }
    async_wait();
    __syncthreads();

    // A fragment (16x32): lane rl holds row, one contiguous 16-run at k = hi*16
    v16bf af;
    {
      const bf16_t* ap = &aS[(w * 16 + rl) * 40];
#pragma unroll
      for (int i = 0; i < 8; ++i) {
        af[i]     = ap[hi * 16 + i];
        af[8 + i] = ap[hi * 16 + 8 + i];
      }
    }

    // 4 B fragments (32x16 each) straight from global W (rows = output cols)
#pragma unroll
    for (int t = 0; t < 4; ++t) {
      const bf16_t* wp = W + (size_t)(n0 + t * 16 + rl) * ldw + k0 + hi * 16;
      v16bf bfv;
#pragma unroll
      for (int i = 0; i < 16; ++i) bfv[i] = wp[i];
      acc[t] = wmma_bf16(af, bfv, acc[t]);
    }
    __syncthreads();
  }

  // store C tile (C layout: VGPR i -> row i + hi*8, col = lane%16)
#pragma unroll
  for (int t = 0; t < 4; ++t) {
#pragma unroll
    for (int i = 0; i < 8; ++i) {
      int row = m0w + i + h8;
      int col = n0 + t * 16 + rl;
      if (BF16OUT)
        ((bf16_t*)C)[(size_t)row * ldc + col] = (bf16_t)acc[t][i];
      else
        ((float*)C)[(size_t)row * ldc + col] = acc[t][i];
    }
  }
}

// ---------------- flash attention (causal) ----------------
// Q,K: (B,T,H,192) bf16 ; V: (B,T,H,128) bf16 ; Oh: (B,T,H,128) bf16
// grid = (B*H, T/128), block = 256 (8 waves, 16 query rows each)
__global__ __launch_bounds__(256, 1)
void mla_flash_attn(const bf16_t* __restrict__ Q, const bf16_t* __restrict__ Kf,
                    const bf16_t* __restrict__ V, bf16_t* __restrict__ Oh) {
  __shared__ bf16_t kS[32 * 200]; // 32 keys x 192 (stride 200)
  __shared__ bf16_t vS[32 * 136]; // 32 keys x 128 (stride 136)
  __shared__ bf16_t pS[128 * 40]; // 128 q-rows x 32 keys (stride 40)

  const int bh = blockIdx.x;
  const int b  = bh / Hh;
  const int h  = bh % Hh;
  const int qs = blockIdx.y * 128;

  const int tid  = threadIdx.x;
  const int w    = tid >> 5;
  const int lane = tid & 31;
  const int rl   = lane & 15;
  const int hi   = lane >> 4;
  const int h8   = hi * 8;

  // Q fragments: 6 slices of K-dim 32 (A layout)
  v16bf qf[6];
  {
    const int tq = qs + w * 16 + rl;
    const bf16_t* qp = Q + ((size_t)(b * Tt + tq) * Hh + h) * QKD;
#pragma unroll
    for (int kk = 0; kk < 6; ++kk)
#pragma unroll
      for (int i = 0; i < 8; ++i) {
        qf[kk][i]     = qp[kk * 32 + hi * 16 + i];
        qf[kk][8 + i] = qp[kk * 32 + hi * 16 + 8 + i];
      }
  }

  v8f o[8];
#pragma unroll
  for (int t = 0; t < 8; ++t) o[t] = v8f_zero();
  float mrow[8], lrow[8];
#pragma unroll
  for (int i = 0; i < 8; ++i) { mrow[i] = -3.0e38f; lrow[i] = 0.0f; }

  const float scale = 0.0721687836f; // 1/sqrt(192)
  const int nkb = (qs + 128) / 32;
  const int key = tid >> 3, seg = tid & 7;

  for (int kb = 0; kb < nkb; ++kb) {
    const int ks0 = kb * 32;
    // K tile: 32 x 192 (each thread async-copies 24 contiguous bf16)
    {
      const bf16_t* src = Kf + ((size_t)(b * Tt + ks0 + key) * Hh + h) * QKD + seg * 24;
      bf16_t* dst = &kS[key * 200 + seg * 24];
      async_copy16(src, dst);
      async_copy16(src + 8, dst + 8);
      async_copy16(src + 16, dst + 16);
    }
    // V tile: 32 x 128 (each thread async-copies 16 contiguous bf16)
    {
      const bf16_t* src = V + ((size_t)(b * Tt + ks0 + key) * Hh + h) * HD + seg * 16;
      bf16_t* dst = &vS[key * 136 + seg * 16];
      async_copy16(src, dst);
      async_copy16(src + 8, dst + 8);
    }
    async_wait();
    __syncthreads();

    // scores: 16 q-rows x 32 keys
    v8f c0 = v8f_zero(), c1 = v8f_zero();
#pragma unroll
    for (int kk = 0; kk < 6; ++kk) {
      v16bf b0, b1;
      const bf16_t* kp0 = &kS[rl * 200 + kk * 32 + hi * 16];
      const bf16_t* kp1 = &kS[(rl + 16) * 200 + kk * 32 + hi * 16];
#pragma unroll
      for (int i = 0; i < 16; ++i) { b0[i] = kp0[i]; b1[i] = kp1[i]; }
      c0 = wmma_bf16(qf[kk], b0, c0);
      c1 = wmma_bf16(qf[kk], b1, c1);
    }

    // causal mask + online softmax update; P into LDS (per-wave region)
    const int tqbase = qs + w * 16 + h8;
#pragma unroll
    for (int i = 0; i < 8; ++i) {
      const int tq = tqbase + i;
      const int k0i = ks0 + rl, k1i = k0i + 16;
      float s0 = c0[i] * scale + (k0i <= tq ? 0.0f : -1.0e9f);
      float s1 = c1[i] * scale + (k1i <= tq ? 0.0f : -1.0e9f);
      float mx = fmaxf(s0, s1);
#pragma unroll
      for (int off = 8; off >= 1; off >>= 1) mx = fmaxf(mx, __shfl_xor(mx, off, 32));
      float mnew = fmaxf(mrow[i], mx);
      float p0 = __expf(s0 - mnew), p1 = __expf(s1 - mnew);
      float ps = p0 + p1;
#pragma unroll
      for (int off = 8; off >= 1; off >>= 1) ps += __shfl_xor(ps, off, 32);
      float corr = __expf(mrow[i] - mnew);
      lrow[i] = lrow[i] * corr + ps;
      mrow[i] = mnew;
#pragma unroll
      for (int t = 0; t < 8; ++t) o[t][i] *= corr;
      pS[(w * 16 + h8 + i) * 40 + rl]      = (bf16_t)p0;
      pS[(w * 16 + h8 + i) * 40 + 16 + rl] = (bf16_t)p1;
    }
    __syncthreads();

    // O += P(16x32) @ V(32x128)
    v16bf pa;
    {
      const bf16_t* pp = &pS[(w * 16 + rl) * 40];
#pragma unroll
      for (int i = 0; i < 8; ++i) {
        pa[i]     = pp[hi * 16 + i];
        pa[8 + i] = pp[hi * 16 + 8 + i];
      }
    }
#pragma unroll
    for (int t = 0; t < 8; ++t) {
      v16bf vb;
#pragma unroll
      for (int i = 0; i < 16; ++i) vb[i] = vS[(hi * 16 + i) * 136 + t * 16 + rl];
      o[t] = wmma_bf16(pa, vb, o[t]);
    }
    __syncthreads();
  }

  // normalize + store
#pragma unroll
  for (int t = 0; t < 8; ++t) {
#pragma unroll
    for (int i = 0; i < 8; ++i) {
      const int tq = qs + w * 16 + h8 + i;
      Oh[((size_t)(b * Tt + tq) * Hh + h) * HD + t * 16 + rl] = (bf16_t)(o[t][i] / lrow[i]);
    }
  }
}

// ---------------- host side ----------------
static void launch_gemm(const bf16_t* A, const bf16_t* W, void* C,
                        int M, int N, int K, bool bf16out, hipStream_t s) {
  dim3 grid(N / 64, M / 128), blk(256);
  if (bf16out)
    gemm_wmma<true><<<grid, blk, 0, s>>>(A, K, W, K, C, N, K);
  else
    gemm_wmma<false><<<grid, blk, 0, s>>>(A, K, W, K, C, N, K);
}

extern "C" void kernel_launch(void* const* d_in, const int* in_sizes, int n_in,
                              void* d_out, int out_size, void* d_ws, size_t ws_size,
                              hipStream_t stream) {
  (void)in_sizes; (void)n_in; (void)out_size; (void)ws_size;
  const float* x    = (const float*)d_in[0];
  const float* pos  = (const float*)d_in[1];
  // d_in[2] = attention_mask: causal mask applied analytically
  const float* Wkv  = (const float*)d_in[3];
  const float* Wkun = (const float*)d_in[4];
  const float* Wvun = (const float*)d_in[5];
  const float* Wqc  = (const float*)d_in[6];
  const float* Wqun = (const float*)d_in[7];
  const float* Wkr  = (const float*)d_in[8];
  const float* Wqr  = (const float*)d_in[9];
  const float* Wout = (const float*)d_in[10];
  float* out = (float*)d_out;

  const int BT = Bb * Tt; // 4096
  char* ws = (char*)d_ws;
  size_t off = 0;
  auto alloc = [&](size_t elems) {
    bf16_t* p = (bf16_t*)(ws + off);
    off += (elems * sizeof(bf16_t) + 255) & ~size_t(255);
    return p;
  };

  bf16_t* xb     = alloc((size_t)BT * Dd);
  bf16_t* wkvb   = alloc((size_t)CKV * Dd);
  bf16_t* wkunb  = alloc((size_t)Dd * CKV);
  bf16_t* wvunb  = alloc((size_t)Dd * CKV);
  bf16_t* wqcb   = alloc((size_t)CQ * Dd);
  bf16_t* wqunb  = alloc((size_t)Dd * CQ);
  bf16_t* wkrb   = alloc((size_t)Rr * Dd);
  bf16_t* wqrb   = alloc((size_t)(Hh * Rr) * CQ);
  bf16_t* woutb  = alloc((size_t)Dd * (Hh * HD));
  bf16_t* kvcb   = alloc((size_t)BT * CKV);
  bf16_t* krb    = alloc((size_t)BT * Rr);
  bf16_t* qcb    = alloc((size_t)BT * CQ);
  bf16_t* fkb    = alloc((size_t)BT * Dd);
  bf16_t* vb     = alloc((size_t)BT * Dd);
  bf16_t* qcontb = alloc((size_t)BT * Dd);
  bf16_t* qropeb = alloc((size_t)BT * Hh * Rr);
  bf16_t* qb     = alloc((size_t)BT * Hh * QKD);
  bf16_t* kb     = alloc((size_t)BT * Hh * QKD);
  bf16_t* ohb    = alloc((size_t)BT * Hh * HD);

  auto cvt = [&](const float* s, bf16_t* d, int n) {
    cvt_f32_bf16<<<(n + 255) / 256, 256, 0, stream>>>(s, d, n);
  };

  // 1) bf16 conversions
  cvt(x, xb, BT * Dd);
  cvt(Wkv, wkvb, CKV * Dd);
  cvt(Wkun, wkunb, Dd * CKV);
  cvt(Wvun, wvunb, Dd * CKV);
  cvt(Wqc, wqcb, CQ * Dd);
  cvt(Wqun, wqunb, Dd * CQ);
  cvt(Wkr, wkrb, Rr * Dd);
  cvt(Wqr, wqrb, Hh * Rr * CQ);
  cvt(Wout, woutb, Dd * Hh * HD);

  // 2) down-projections from x
  launch_gemm(xb, wkvb, kvcb, BT, CKV, Dd, true, stream);  // kv_c
  launch_gemm(xb, wkrb, krb, BT, Rr, Dd, true, stream);    // k_rope pre
  launch_gemm(xb, wqcb, qcb, BT, CQ, Dd, true, stream);    // q_c

  // 3) rope on k_rope (H=1)
  {
    int n = BT * 1 * (Rr / 2);
    rope_bf16<<<(n + 255) / 256, 256, 0, stream>>>(krb, pos, 1, Rr / 2, n);
  }

  // 4) up-projections
  launch_gemm(kvcb, wkunb, fkb, BT, Dd, CKV, true, stream);      // full_k
  launch_gemm(kvcb, wvunb, vb, BT, Dd, CKV, true, stream);       // full_v
  launch_gemm(qcb, wqunb, qcontb, BT, Dd, CQ, true, stream);     // q_content
  launch_gemm(qcb, wqrb, qropeb, BT, Hh * Rr, CQ, true, stream); // q_rope pre

  // 5) rope on q_rope (H=16)
  {
    int n = BT * Hh * (Rr / 2);
    rope_bf16<<<(n + 255) / 256, 256, 0, stream>>>(qropeb, pos, Hh, Rr / 2, n);
  }

  // 6) concat heads
  {
    int n = BT * Hh * QKD;
    pack_q_kernel<<<(n + 255) / 256, 256, 0, stream>>>(qcontb, qropeb, qb, n);
    pack_k_kernel<<<(n + 255) / 256, 256, 0, stream>>>(fkb, krb, kb, n);
  }

  // 7) causal flash attention
  {
    dim3 grid(Bb * Hh, Tt / 128), blk(256);
    mla_flash_attn<<<grid, blk, 0, stream>>>(qb, kb, vb, ohb);
  }

  // 8) output projection (fp32 out)
  launch_gemm(ohb, woutb, out, BT, Dd, Hh * HD, false, stream);
}
